// LinearAttention_42760694399116
// MI455X (gfx1250) — compile-verified, matrix operand order checked by reading, not run on hardware
//
#include <hip/hip_runtime.h>

#define HEADS 8
#define DIMH  64
#define NB    8
#define NC    512
#define NL    8192

typedef __attribute__((ext_vector_type(16))) _Float16 v16h;
typedef __attribute__((ext_vector_type(8)))  _Float16 v8h;
typedef __attribute__((ext_vector_type(8)))  float    v8f;

// ---- CDNA5 async global->LDS copy (16 bytes per lane), tracked by ASYNCcnt.
__device__ __forceinline__ void async_copy16(const _Float16* gsrc, _Float16* lds_dst) {
  // Low 32 bits of a generic pointer to LDS are the wave-relative LDS byte offset.
  unsigned lds_off = (unsigned)(uintptr_t)lds_dst;
  asm volatile("global_load_async_to_lds_b128 %0, %1, off"
               :: "v"(lds_off), "v"(gsrc) : "memory");
}
__device__ __forceinline__ void wait_async_le1() {
  asm volatile("s_wait_asynccnt 0x1" ::: "memory");
}
__device__ __forceinline__ void wait_async_le0() {
  asm volatile("s_wait_asynccnt 0x0" ::: "memory");
}

// Load one 16x32 f16 fragment (A row-major by M, or B column-major by N) from
// memory where K is contiguous within each row and rows are `stride` halves apart.
// Per ISA 7.12.2: lane r=lane&15 selects the row; lanes 16-31 hold K offset +8;
// halves [0..7] cover K = 8g..8g+7, halves [8..15] cover K = 16+8g..16+8g+7.
// Works for global (global_load_b128) and LDS (ds_load_b128) pointers.
__device__ __forceinline__ v16h load_frag(const _Float16* __restrict__ base,
                                          int stride, int lane) {
  const int r = lane & 15;
  const int g = (lane >> 4) << 3;   // 0 or 8
  const _Float16* p = base + (size_t)r * (size_t)stride + g;
  v8h lo = *(const v8h*)(p);
  v8h hi = *(const v8h*)(p + 16);
  return __builtin_shufflevector(lo, hi, 0,1,2,3,4,5,6,7,8,9,10,11,12,13,14,15);
}

__device__ __forceinline__ v8f wmma_f16(v16h a, v16h b, v8f c) {
  return __builtin_amdgcn_wmma_f32_16x16x32_f16(false, a, false, b, (short)0, c,
                                                false, false);
}

// LDS B-tile: 64 rows x 32 halves, padded to 40 halves/row (80B: conflict-free
// ds_load_b128 since 20*row mod 64 banks is distinct for row 0..15).
#define BROW 40

// One K=32 step of a 16x64 output strip: A from global (L2-hot weights),
// four B fragments from the LDS tile.
__device__ __forceinline__ void gemm_step(v8f (&acc)[4], const _Float16* __restrict__ abase,
                                          int kk, int astride,
                                          const _Float16* __restrict__ lb, int lane) {
  v16h af = load_frag(abase + kk, astride, lane);   // global_load_b128 x2
#pragma unroll
  for (int j = 0; j < 4; ++j) {
    v16h bf = load_frag(lb + j * 16 * BROW, BROW, lane);  // ds_load_b128 x2
    acc[j] = wmma_f16(af, bf, acc[j]);
  }
}

// ---------------- Kernel 0a: x[b][c][l] f32 -> xT[b][l][c] f16 (LDS tile transpose)
__global__ void k_transpose_x(const float* __restrict__ x, _Float16* __restrict__ xT) {
  __shared__ _Float16 tile[32][33];
  const int b  = blockIdx.z;
  const int c0 = blockIdx.y * 32;
  const int l0 = blockIdx.x * 32;
  const int t  = threadIdx.x; // 256
  {
    const int cl = t >> 3;
    const int lq = (t & 7) * 4;
    const float4 v = *(const float4*)(x + ((size_t)b * NC + c0 + cl) * NL + l0 + lq);
    tile[cl][lq + 0] = (_Float16)v.x;
    tile[cl][lq + 1] = (_Float16)v.y;
    tile[cl][lq + 2] = (_Float16)v.z;
    tile[cl][lq + 3] = (_Float16)v.w;
  }
  __syncthreads();
  {
    const int ll = t >> 3;
    const int cq = (t & 7) * 4;
    _Float16* dst = xT + ((size_t)b * NL + l0 + ll) * NC + c0 + cq;
    dst[0] = tile[cq + 0][ll];
    dst[1] = tile[cq + 1][ll];
    dst[2] = tile[cq + 2][ll];
    dst[3] = tile[cq + 3][ll];
  }
}

// ---------------- Kernel 0b: weight f32 -> f16
__global__ void k_convert_w(const float* __restrict__ wq, const float* __restrict__ wo,
                            _Float16* __restrict__ wqh, _Float16* __restrict__ woh) {
  const int i = blockIdx.x * 256 + threadIdx.x;
  if (i < 3 * NC * NC) wqh[i] = (_Float16)wq[i];
  if (i < NC * NC)     woh[i] = (_Float16)wo[i];
}

// ---------------- Kernel 1: qkv[o,l] = W_qkv[o,:] . x[b,:,l]   (per batch)
// B tile (64 l x 32 c) staged in LDS with async copies, symmetric double-buffer
// software pipeline. q -> [b][h][l][d], k/v -> [b][h][d][l], all f16.
__global__ void __launch_bounds__(256) k_qkv(const _Float16* __restrict__ wq,
                                             const _Float16* __restrict__ xT,
                                             _Float16* __restrict__ qb,
                                             _Float16* __restrict__ kb,
                                             _Float16* __restrict__ vb) {
  __shared__ __align__(16) _Float16 bt[2][64][BROW];
  const int t    = threadIdx.x;
  const int lane = t & 31;
  const int w    = t >> 5;                      // 8 waves
  const int b    = blockIdx.z;
  const int o0   = (blockIdx.y * 8 + w) * 16;   // M tile (0..1520)
  const int l0   = blockIdx.x * 64;             // 4 N tiles

  const _Float16* abase = wq + (size_t)o0 * NC;
  // Async staging assignment: thread t copies 16B = row t>>2, halves (t&3)*8.
  const int srow   = t >> 2;
  const int schunk = (t & 3) * 8;
  const _Float16* gsrc = xT + ((size_t)b * NL + l0 + srow) * NC + schunk;
  _Float16* l0dst = &bt[0][srow][schunk];
  _Float16* l1dst = &bt[1][srow][schunk];

  async_copy16(gsrc, l0dst);              // S0 -> buf0
  async_copy16(gsrc + 32, l1dst);         // S1 -> buf1

  v8f acc[4] = {};
  for (int s = 0; s < 14; s += 2) {       // symmetric: no acc phi rotation
    wait_async_le1(); __syncthreads();
    gemm_step(acc, abase, s * 32, NC, &bt[0][0][0], lane);
    __syncthreads();
    async_copy16(gsrc + (s + 2) * 32, l0dst);
    wait_async_le1(); __syncthreads();
    gemm_step(acc, abase, (s + 1) * 32, NC, &bt[1][0][0], lane);
    __syncthreads();
    async_copy16(gsrc + (s + 3) * 32, l1dst);
  }
  wait_async_le1(); __syncthreads();
  gemm_step(acc, abase, 14 * 32, NC, &bt[0][0][0], lane);
  wait_async_le0(); __syncthreads();
  gemm_step(acc, abase, 15 * 32, NC, &bt[1][0][0], lane);

  const int g = lane >> 4, n = lane & 15;
#pragma unroll
  for (int j = 0; j < 4; ++j) {
    const int l = l0 + j * 16 + n;
#pragma unroll
    for (int r = 0; r < 8; ++r) {
      const int o = o0 + r + g * 8;
      const float val = acc[j][r];
      if (o < 512) {
        const int h = o >> 6, d = o & 63;
        qb[(((size_t)b * HEADS + h) * NL + l) * DIMH + d] = (_Float16)val;
      } else if (o < 1024) {
        const int o2 = o - 512, h = o2 >> 6, d = o2 & 63;
        kb[(((size_t)b * HEADS + h) * DIMH + d) * NL + l] = (_Float16)val;
      } else {
        const int o2 = o - 1024, h = o2 >> 6, d = o2 & 63;
        vb[(((size_t)b * HEADS + h) * DIMH + d) * NL + l] = (_Float16)val;
      }
    }
  }
}

// ---------------- Kernel 2: softmax over l for each (b,h,d) row of k, in place
__global__ void k_softmax(_Float16* __restrict__ kb) {
  __shared__ float red[256];
  const int t = threadIdx.x;
  _Float16* p = kb + (size_t)blockIdx.x * NL + t * 32;   // 32 contiguous per thread
  v8h ch[4];
  float vals[32];
  float m = -3.0e38f;
#pragma unroll
  for (int c = 0; c < 4; ++c) ch[c] = *(const v8h*)(p + c * 8);  // b128 loads
#pragma unroll
  for (int i = 0; i < 32; ++i) { vals[i] = (float)ch[i >> 3][i & 7]; m = fmaxf(m, vals[i]); }
  red[t] = m; __syncthreads();
  for (int s = 128; s > 0; s >>= 1) { if (t < s) red[t] = fmaxf(red[t], red[t + s]); __syncthreads(); }
  m = red[0]; __syncthreads();
  float sum = 0.f;
#pragma unroll
  for (int i = 0; i < 32; ++i) { vals[i] = __expf(vals[i] - m); sum += vals[i]; }
  red[t] = sum; __syncthreads();
  for (int s = 128; s > 0; s >>= 1) { if (t < s) red[t] += red[t + s]; __syncthreads(); }
  const float inv = 1.0f / red[0];
#pragma unroll
  for (int i = 0; i < 32; ++i) ch[i >> 3][i & 7] = (_Float16)(vals[i] * inv);
#pragma unroll
  for (int c = 0; c < 4; ++c) *(v8h*)(p + c * 8) = ch[c];        // b128 stores
}

// ---------------- Kernel 3: CT[b,h][e][d] = sum_l v[e,l] * k_sm[d,l]
// Register double-buffered so WMMAs overlap the next fragment loads.
__global__ void k_context(const _Float16* __restrict__ vb, const _Float16* __restrict__ kb,
                          _Float16* __restrict__ ct) {
  const int lane = threadIdx.x & 31;
  const int w    = threadIdx.x >> 5;  // 16 waves -> 4x4 tiles of 64x64
  const int bh   = blockIdx.x;
  const int et   = w & 3, dt = w >> 2;
  const _Float16* abase = vb + ((size_t)bh * DIMH + et * 16) * NL;
  const _Float16* bbase = kb + ((size_t)bh * DIMH + dt * 16) * NL;
  v8f acc = {};
  v16h af = load_frag(abase, NL, lane);
  v16h bf = load_frag(bbase, NL, lane);
  for (int kk = 32; kk < NL; kk += 32) {
    v16h af2 = load_frag(abase + kk, NL, lane);
    v16h bf2 = load_frag(bbase + kk, NL, lane);
    __builtin_prefetch(abase + kk + 256, 0, 3);
    acc = wmma_f16(af, bf, acc);
    af = af2; bf = bf2;
  }
  acc = wmma_f16(af, bf, acc);
  const int g = lane >> 4, n = lane & 15;
#pragma unroll
  for (int r = 0; r < 8; ++r) {
    const int e = et * 16 + r + g * 8;
    const int d = dt * 16 + n;
    ct[(size_t)bh * 4096 + e * 64 + d] = (_Float16)acc[r];
  }
}

// ---------------- Kernel 4: out[b][l][h*64+e] = sum_d CT[e,d] * q[b,h,l,d]
__global__ void k_out(const _Float16* __restrict__ ct, const _Float16* __restrict__ qb,
                      _Float16* __restrict__ ob) {
  const int lane = threadIdx.x & 31;
  const int et   = threadIdx.x >> 5;  // 4 waves = 4 M-tiles
  const int bh   = blockIdx.y;
  const int l0   = blockIdx.x * 64;
  const _Float16* abase = ct + (size_t)bh * 4096 + et * 16 * 64;
  v16h a0 = load_frag(abase + 0, 64, lane);
  v16h a1 = load_frag(abase + 32, 64, lane);
  const int b = bh >> 3, h = bh & 7;
  const int g = lane >> 4, n = lane & 15;
#pragma unroll
  for (int j = 0; j < 4; ++j) {
    const _Float16* bbase = qb + ((size_t)bh * NL + l0 + j * 16) * DIMH;
    v16h b0 = load_frag(bbase + 0, 64, lane);
    v16h b1 = load_frag(bbase + 32, 64, lane);
    v8f acc = {};
    acc = wmma_f16(a0, b0, acc);
    acc = wmma_f16(a1, b1, acc);
    const int l = l0 + j * 16 + n;
#pragma unroll
    for (int r = 0; r < 8; ++r) {
      const int e = et * 16 + r + g * 8;
      ob[((size_t)b * NL + l) * 512 + h * 64 + e] = (_Float16)acc[r];
    }
  }
}

// ---------------- Kernel 5: y[b,c,l] = W_out[c,:] . out[b,l,:] + bias[c]   (f32 out)
// Same async-LDS software pipeline as k_qkv.
__global__ void __launch_bounds__(256) k_final(const _Float16* __restrict__ wo,
                                               const _Float16* __restrict__ ob,
                                               const float* __restrict__ bias,
                                               float* __restrict__ y) {
  __shared__ __align__(16) _Float16 bt[2][64][BROW];
  const int t    = threadIdx.x;
  const int lane = t & 31;
  const int w    = t >> 5;
  const int b    = blockIdx.z;
  const int c0   = (blockIdx.y * 8 + w) * 16;
  const int l0   = blockIdx.x * 64;
  const _Float16* abase = wo + (size_t)c0 * 512;
  const int srow   = t >> 2;
  const int schunk = (t & 3) * 8;
  const _Float16* gsrc = ob + ((size_t)b * NL + l0 + srow) * 512 + schunk;
  _Float16* l0dst = &bt[0][srow][schunk];
  _Float16* l1dst = &bt[1][srow][schunk];

  async_copy16(gsrc, l0dst);
  async_copy16(gsrc + 32, l1dst);

  v8f acc[4] = {};
  for (int s = 0; s < 14; s += 2) {
    wait_async_le1(); __syncthreads();
    gemm_step(acc, abase, s * 32, 512, &bt[0][0][0], lane);
    __syncthreads();
    async_copy16(gsrc + (s + 2) * 32, l0dst);
    wait_async_le1(); __syncthreads();
    gemm_step(acc, abase, (s + 1) * 32, 512, &bt[1][0][0], lane);
    __syncthreads();
    async_copy16(gsrc + (s + 3) * 32, l1dst);
  }
  wait_async_le1(); __syncthreads();
  gemm_step(acc, abase, 14 * 32, 512, &bt[0][0][0], lane);
  wait_async_le0(); __syncthreads();
  gemm_step(acc, abase, 15 * 32, 512, &bt[1][0][0], lane);

  const int g = lane >> 4, n = lane & 15;
#pragma unroll
  for (int j = 0; j < 4; ++j) {
    const int l = l0 + j * 16 + n;
#pragma unroll
    for (int r = 0; r < 8; ++r) {
      const int c = c0 + r + g * 8;
      y[((size_t)b * NC + c) * NL + l] = acc[j][r] + bias[c];
    }
  }
}

extern "C" void kernel_launch(void* const* d_in, const int* in_sizes, int n_in,
                              void* d_out, int out_size, void* d_ws, size_t ws_size,
                              hipStream_t stream) {
  const float* x  = (const float*)d_in[0];
  const float* wq = (const float*)d_in[1];
  const float* wo = (const float*)d_in[2];
  const float* bo = (const float*)d_in[3];
  float* y = (float*)d_out;
  char* ws = (char*)d_ws;

  const size_t SZ_XT = (size_t)NB * NL * NC * 2;            // 64 MiB
  const size_t SZ_WQ = (size_t)3 * NC * NC * 2;             // 1.5 MiB
  const size_t SZ_WO = (size_t)NC * NC * 2;                 // 0.5 MiB
  const size_t SZ_Q  = (size_t)NB * HEADS * NL * DIMH * 2;  // 64 MiB
  const size_t SZ_CT = (size_t)NB * HEADS * DIMH * DIMH * 2;

  size_t off = 0;
  _Float16* xT  = (_Float16*)(ws + off); off += SZ_XT;
  _Float16* wqh = (_Float16*)(ws + off); off += SZ_WQ;
  _Float16* woh = (_Float16*)(ws + off); off += SZ_WO;
  _Float16* qb  = (_Float16*)(ws + off); off += SZ_Q;
  _Float16* kb  = (_Float16*)(ws + off); off += SZ_Q;
  _Float16* vb  = (_Float16*)(ws + off); off += SZ_Q;
  _Float16* ctb = (_Float16*)(ws + off); off += SZ_CT;
  _Float16* ob  = (_Float16*)(ws + off); off += SZ_Q;
  (void)off; (void)ws_size; (void)in_sizes; (void)n_in; (void)out_size;

  k_transpose_x<<<dim3(NL / 32, NC / 32, NB), 256, 0, stream>>>(x, xT);
  k_convert_w  <<<dim3((3 * NC * NC) / 256), 256, 0, stream>>>(wq, wo, wqh, woh);
  k_qkv        <<<dim3(NL / 64, 12, NB), 256, 0, stream>>>(wqh, xT, qb, kb, vb);
  k_softmax    <<<dim3(NB * HEADS * DIMH), 256, 0, stream>>>(kb);
  k_context    <<<dim3(NB * HEADS), 512, 0, stream>>>(vb, kb, ctb);
  k_out        <<<dim3(NL / 64, NB * HEADS), 128, 0, stream>>>(ctb, qb, ob);
  k_final      <<<dim3(NL / 64, 4, NB), 256, 0, stream>>>(woh, ob, bo, y);
}
